// Model_39676907881850
// MI455X (gfx1250) — matrix operand order repaired and checked by reading, not compiled, not run on hardware
//
#include <hip/hip_runtime.h>

typedef __attribute__((ext_vector_type(16))) _Float16 v16h;
typedef __attribute__((ext_vector_type(8)))  _Float16 v8h;
typedef __attribute__((ext_vector_type(4)))  _Float16 v4h;
typedef __attribute__((ext_vector_type(2)))  _Float16 v2h;
typedef __attribute__((ext_vector_type(8)))  float    v8f;

#define NN  128
#define DD  32
#define KK  4096   // NN*DD
#define BK  64     // phase-1 staged K-chunk
#define LDA 88     // phase-1 LDS row stride (halves): 176B, 16B-aligned, conflict-free
#define LDT 136    // phase-2 LDS row stride (halves): 272B, 16B-aligned

// ---- CDNA5 async memory->LDS copy (ASYNCcnt path, §15.18.3 op 98) -----------
static __device__ __forceinline__ void async_ld_b128(unsigned lds_off, const void* g) {
  asm volatile("global_load_async_to_lds_b128 %0, %1, off"
               :: "v"(lds_off), "v"(g) : "memory");
}
static __device__ __forceinline__ void wait_async0() {
  asm volatile("s_wait_asynccnt 0x0" ::: "memory");
}

// ---------------------------------------------------------------------------
// Phase 1: scores[j] = (a[j] @ a[j]^T)/64 ; probs[j] = softmax(scores + mask)
// Double-buffered LDS; next chunk's global loads issued before the WMMA burst
// and landed after it.  A lane(h,r): row m0+r, contiguous chunk {k0+16h .. +15};
// B lane(h,r): col n0+r, contiguous chunk {k0+16h .. +15}.
// ---------------------------------------------------------------------------
__global__ __launch_bounds__(256) void qk_softmax_f16(
    const float* __restrict__ x1, const float* __restrict__ mask,
    _Float16* __restrict__ probs)
{
  __shared__ _Float16 As[2][NN * LDA];          // 2 x 22 KB

  const int j    = blockIdx.x;
  const int tid  = threadIdx.x;
  const int wav  = tid >> 5;
  const int lane = tid & 31;
  const int h    = lane >> 4;
  const int r    = lane & 15;
  const int m0   = wav * 16;

  const float* aj = x1 + (size_t)j * (size_t)(NN * KK);

  // prologue: stage chunk 0 (batched loads -> one wait -> stores)
  {
    float4 f[8];
    #pragma unroll
    for (int p = 0; p < 8; ++p) {
      int li = p * 1024 + tid * 4, row = li >> 6, col = li & 63;
      f[p] = *(const float4*)(aj + (size_t)row * KK + col);
    }
    #pragma unroll
    for (int p = 0; p < 8; ++p) {
      int li = p * 1024 + tid * 4, row = li >> 6, col = li & 63;
      v4h pk = { (_Float16)f[p].x, (_Float16)f[p].y, (_Float16)f[p].z, (_Float16)f[p].w };
      *(v4h*)(&As[0][row * LDA + col]) = pk;
    }
  }
  __syncthreads();

  v8f acc[8] = {};
  int cur = 0;
  for (int kt = 0; kt < KK; kt += BK) {
    const bool more = (kt + BK) < KK;
    float4 f[8];
    if (more) {                                   // issue next chunk, no wait yet
      #pragma unroll
      for (int p = 0; p < 8; ++p) {
        int li = p * 1024 + tid * 4, row = li >> 6, col = li & 63;
        f[p] = *(const float4*)(aj + (size_t)row * KK + (kt + BK) + col);
      }
    }

    // ---- WMMA burst on As[cur] ----
    const _Float16* Ab = &As[cur][(m0 + r) * LDA];
    v16h aF0, aF1;
    *(v8h*)&aF0       = *(const v8h*)(Ab + 16 * h);
    *((v8h*)&aF0 + 1) = *(const v8h*)(Ab + 8 + 16 * h);
    *(v8h*)&aF1       = *(const v8h*)(Ab + 32 + 16 * h);
    *((v8h*)&aF1 + 1) = *(const v8h*)(Ab + 40 + 16 * h);
    #pragma unroll
    for (int t = 0; t < 8; ++t) {
      const _Float16* Bb = &As[cur][(t * 16 + r) * LDA + 16 * h];
      v16h b0, b1;
      *(v8h*)&b0       = *(const v8h*)(Bb);
      *((v8h*)&b0 + 1) = *(const v8h*)(Bb + 8);
      *(v8h*)&b1       = *(const v8h*)(Bb + 32);
      *((v8h*)&b1 + 1) = *(const v8h*)(Bb + 40);
      acc[t] = __builtin_amdgcn_wmma_f32_16x16x32_f16(
          false, aF0, false, b0, (short)0, acc[t], false, false);
      acc[t] = __builtin_amdgcn_wmma_f32_16x16x32_f16(
          false, aF1, false, b1, (short)0, acc[t], false, false);
    }

    if (more) {                                   // land loads, fill other buffer
      #pragma unroll
      for (int p = 0; p < 8; ++p) {
        int li = p * 1024 + tid * 4, row = li >> 6, col = li & 63;
        v4h pk = { (_Float16)f[p].x, (_Float16)f[p].y, (_Float16)f[p].z, (_Float16)f[p].w };
        *(v4h*)(&As[cur ^ 1][row * LDA + col]) = pk;
      }
    }
    __syncthreads();
    cur ^= 1;
  }

  // Masked row-softmax.  acc[t][v] = scores[m0+v+8h][t*16+r]; rows live in one
  // 16-lane half -> shfl_xor 8/4/2/1 stays in-half.
  const float scale = 0.015625f;                  // 1/sqrt(4096)
  #pragma unroll
  for (int v = 0; v < 8; ++v) {
    const int srow = m0 + v + 8 * h;
    const float* mrow = mask + srow * NN;
    float x[8];
    float mx = -3.402823466e38f;
    #pragma unroll
    for (int t = 0; t < 8; ++t) {
      x[t] = acc[t][v] * scale + mrow[t * 16 + r];
      mx = fmaxf(mx, x[t]);
    }
    #pragma unroll
    for (int off = 8; off > 0; off >>= 1)
      mx = fmaxf(mx, __shfl_xor(mx, off, 32));
    float s = 0.f;
    #pragma unroll
    for (int t = 0; t < 8; ++t) { x[t] = __expf(x[t] - mx); s += x[t]; }
    #pragma unroll
    for (int off = 8; off > 0; off >>= 1)
      s += __shfl_xor(s, off, 32);
    const float inv = 1.0f / s;
    _Float16* prow = probs + ((size_t)j * NN + srow) * NN;
    #pragma unroll
    for (int t = 0; t < 8; ++t)
      prow[t * 16 + r] = (_Float16)(x[t] * inv);
  }
}

// ---------------------------------------------------------------------------
// Phase 2: out[i,j] = probs[j] @ x1[i,j].  Block = (j, 8 i's): probs[j] staged
// once via ASYNC copies, x1 tiles double-buffered & transposed in LDS.
// ---------------------------------------------------------------------------
__global__ __launch_bounds__(256) void pv_f16(
    const float* __restrict__ x1, const _Float16* __restrict__ probs,
    float* __restrict__ out)
{
  __shared__ _Float16 Ps[NN * LDT];               // probs[j]        (~34.8 KB)
  __shared__ _Float16 Xs[2][DD * LDT];            // x1[i,j]^T       (2 x 8.7 KB)

  const int ig   = blockIdx.x;
  const int j    = blockIdx.y;
  const int tid  = threadIdx.x;
  const int wav  = tid >> 5;
  const int lane = tid & 31;
  const int h    = lane >> 4;
  const int r    = lane & 15;
  const int m0   = wav * 16;

  // stage probs[j] with async memory->LDS copies (pure f16 copy, no convert)
  const _Float16* pj = probs + (size_t)j * (NN * NN);
  #pragma unroll
  for (int p = 0; p < 8; ++p) {
    int li = p * 2048 + tid * 8, s = li >> 7, k = li & 127;
    async_ld_b128((unsigned)(size_t)(Ps + s * LDT + k), pj + li);
  }

  // prefetch + stage x1 tile for ii=0 (coalesced 128B rows, transposed store)
  const int dcol = tid & 31;
  const int kb   = (tid >> 5) * 16;
  float f[16];
  {
    const float* x0 = x1 + ((size_t)(ig * 8) * NN + j) * (size_t)KK;
    #pragma unroll
    for (int q = 0; q < 16; ++q) f[q] = x0[(size_t)(kb + q) * DD + dcol];
    #pragma unroll
    for (int q = 0; q < 16; q += 2) {
      v2h pk = { (_Float16)f[q], (_Float16)f[q + 1] };
      *(v2h*)(&Xs[0][dcol * LDT + kb + q]) = pk;
    }
  }
  wait_async0();

  int cur = 0;
  for (int ii = 0; ii < 8; ++ii) {
    __syncthreads();                              // Ps/Xs[cur] visible, prev compute done
    const bool more = ii < 7;
    if (more) {                                   // issue next tile's loads, no wait
      const float* xn = x1 + ((size_t)(ig * 8 + ii + 1) * NN + j) * (size_t)KK;
      #pragma unroll
      for (int q = 0; q < 16; ++q) f[q] = xn[(size_t)(kb + q) * DD + dcol];
    }

    // ---- WMMA burst: hoisted A frags, per-tile B burst ----
    v8f acc[2] = {};
    const _Float16* Ab = Ps + (m0 + r) * LDT;
    v16h aF[4];
    #pragma unroll
    for (int kq = 0; kq < 4; ++kq) {
      *(v8h*)&aF[kq]       = *(const v8h*)(Ab + kq * 32 + 16 * h);
      *((v8h*)&aF[kq] + 1) = *(const v8h*)(Ab + kq * 32 + 8 + 16 * h);
    }
    #pragma unroll
    for (int t = 0; t < 2; ++t) {
      const _Float16* Bb = &Xs[cur][(t * 16 + r) * LDT + 16 * h];
      v16h bF[4];
      #pragma unroll
      for (int kq = 0; kq < 4; ++kq) {
        *(v8h*)&bF[kq]       = *(const v8h*)(Bb + kq * 32);
        *((v8h*)&bF[kq] + 1) = *(const v8h*)(Bb + kq * 32 + 8);
      }
      #pragma unroll
      for (int kq = 0; kq < 4; ++kq)
        acc[t] = __builtin_amdgcn_wmma_f32_16x16x32_f16(
            false, aF[kq], false, bF[kq], (short)0, acc[t], false, false);
    }

    const int i = ig * 8 + ii;
    float* obase = out + (((size_t)i * NN + j) * NN) * DD;
    #pragma unroll
    for (int t = 0; t < 2; ++t)
      #pragma unroll
      for (int v = 0; v < 8; ++v)
        obase[(size_t)(m0 + v + 8 * h) * DD + t * 16 + r] = acc[t][v];

    if (more) {                                   // land loads into other buffer
      #pragma unroll
      for (int q = 0; q < 16; q += 2) {
        v2h pk = { (_Float16)f[q], (_Float16)f[q + 1] };
        *(v2h*)(&Xs[cur ^ 1][dcol * LDT + kb + q]) = pk;
      }
    }
    cur ^= 1;
  }
}

extern "C" void kernel_launch(void* const* d_in, const int* in_sizes, int n_in,
                              void* d_out, int out_size, void* d_ws, size_t ws_size,
                              hipStream_t stream) {
  const float* x1   = (const float*)d_in[0];   // [128,128,128,32] fp32
  const float* mask = (const float*)d_in[1];   // [128,128] fp32
  _Float16* probs   = (_Float16*)d_ws;         // 128^3 * 2B = 4 MiB scratch
  float* out        = (float*)d_out;           // [128,128,128,32] fp32

  qk_softmax_f16<<<dim3(NN), dim3(256), 0, stream>>>(x1, mask, probs);
  pv_f16<<<dim3(16, NN), dim3(256), 0, stream>>>(x1, probs, out);
}
